// MultiHeadSelfAttention_56753697849737
// MI455X (gfx1250) — compile-verified, matrix operand order checked by reading, not run on hardware
//
#include <hip/hip_runtime.h>
#include <hip/hip_bf16.h>
#include <stdint.h>

// ---------------------------------------------------------------------------
// MHA forward for B=2, S=2048, E=1024, H=16, D=64 on gfx1250 (CDNA5, wave32).
// Matrix math: v_wmma_f32_16x16x32_bf16. Attention K/V tiles staged with
// global_load_async_to_lds_b128 (ASYNCcnt). Fragments loaded as ds_load_b128.
// ---------------------------------------------------------------------------

#define B_   2
#define S_   2048
#define E_   1024
#define H_   16
#define D_   64

typedef __attribute__((ext_vector_type(16))) __bf16 v16bf;
typedef __attribute__((ext_vector_type(8)))  float  v8f;

union Frag16 { v16bf v; uint32_t u[8]; };

// float -> bf16 (round to nearest even), raw u16
__device__ __forceinline__ uint16_t f2bf(float f) {
    uint32_t x = __float_as_uint(f);
    return (uint16_t)((x + 0x7FFFu + ((x >> 16) & 1u)) >> 16);
}
__device__ __forceinline__ uint32_t pack2bf(float lo, float hi) {
    return (uint32_t)f2bf(lo) | ((uint32_t)f2bf(hi) << 16);
}

// 16-bit A/B fragment (ISA 7.12.2): per lane the 16 halves live at dwords
// {0..3}+4*half and {8..11}+4*half of a 16-dword (32-element) K-chunk row.
// Rows are 16B-aligned -> two ds_load_b128 per fragment.
__device__ __forceinline__ void load_frag(Frag16& f, const uint32_t* row, int half) {
    *(uint4*)&f.u[0] = *(const uint4*)(row + 4 * half);
    *(uint4*)&f.u[4] = *(const uint4*)(row + 8 + 4 * half);
}

__device__ __forceinline__ v8f wmma_bf16(const Frag16& a, const Frag16& b, v8f c) {
    return __builtin_amdgcn_wmma_f32_16x16x32_bf16(
        false, a.v, false, b.v, (short)0, c, false, false);
}

// async 16B global -> LDS copy (per active lane), tracked by ASYNCcnt
__device__ __forceinline__ void async_g2l_b128(uint32_t lds_off, const void* gptr) {
    asm volatile("global_load_async_to_lds_b128 %0, %1, off"
                 :: "v"(lds_off), "v"((uint64_t)(uintptr_t)gptr) : "memory");
}
__device__ __forceinline__ void wait_async0() {
    asm volatile("s_wait_asynccnt 0x0" ::: "memory");
}
__device__ __forceinline__ uint32_t lds_off_of(const void* p) {
    return (uint32_t)(uintptr_t)p;  // flat LDS aperture: low 32 bits = LDS addr
}

// ---------------------------------------------------------------------------
// Generic tiled GEMM:  Out = A(MxK) * W(KxN) + bias,  A fp32 or bf16, W fp32.
// Block = 64x64 tile, 128 threads (4 waves 2x2), wave = 32x32 (4 wmma accs).
// mode 0: bf16 -> [B,H,S,D] (Q,K)   mode 1: bf16 -> [B,H,D,S] (V, transposed)
// mode 2: fp32 -> [M,N] (output projection)
// ---------------------------------------------------------------------------
template <bool ABF16>
__global__ __launch_bounds__(128)
void gemm_bf16_kernel(const void* __restrict__ Aptr,
                      const float* __restrict__ W,
                      const float* __restrict__ bias,
                      void* __restrict__ Out,
                      int M, int N, int K, int lda, int ldb, int mode) {
    __shared__ uint32_t lds_a[64][20];  // 64 rows x 32 bf16 (16 dw) + pad, 16B-aligned rows
    __shared__ uint32_t lds_b[64][20];  // 64 cols x 32 bf16 (16 dw) + pad

    const int tid    = threadIdx.x;
    const int lane   = tid & 31;
    const int wave   = tid >> 5;
    const int half   = lane >> 4;
    const int lane16 = lane & 15;
    const int wm     = (wave >> 1) * 32;
    const int wn     = (wave & 1) * 32;
    const int m0     = blockIdx.y * 64;
    const int n0     = blockIdx.x * 64;

    v8f acc[2][2] = {};

    for (int k0 = 0; k0 < K; k0 += 32) {
        __syncthreads();
        // ---- stage A tile (64 x 32) as bf16, b128 global loads ----
        if (ABF16) {
            const uint4* A4 = (const uint4*)Aptr;  // 8 bf16 per load
            for (int idx = tid; idx < 64 * 4; idx += 128) {
                const int r = idx >> 2, c = idx & 3;
                *(uint4*)&lds_a[r][c * 4] =
                    A4[(((size_t)(m0 + r) * lda + k0) >> 3) + c];
            }
        } else {
            const float4* A4 = (const float4*)Aptr;
            for (int idx = tid; idx < 64 * 8; idx += 128) {
                const int r = idx >> 3, c4 = idx & 7;
                const float4 v = A4[(((size_t)(m0 + r) * lda + k0) >> 2) + c4];
                lds_a[r][c4 * 2]     = pack2bf(v.x, v.y);
                lds_a[r][c4 * 2 + 1] = pack2bf(v.z, v.w);
            }
        }
        // ---- stage B tile transposed (lds_b[n][k]), b128 row loads ----
        {
            uint16_t* lb = (uint16_t*)&lds_b[0][0];  // row stride 40 halves
            for (int idx = tid; idx < 32 * 16; idx += 128) {
                const int k = idx >> 4, n4 = idx & 15;
                const float4 w = *(const float4*)&W[(size_t)(k0 + k) * ldb + n0 + n4 * 4];
                lb[(n4 * 4 + 0) * 40 + k] = f2bf(w.x);
                lb[(n4 * 4 + 1) * 40 + k] = f2bf(w.y);
                lb[(n4 * 4 + 2) * 40 + k] = f2bf(w.z);
                lb[(n4 * 4 + 3) * 40 + k] = f2bf(w.w);
            }
        }
        // prefetch next K-chunk (global_prefetch_b8)
        if (k0 + 32 < K) {
            if (ABF16)
                __builtin_prefetch((const uint16_t*)Aptr +
                                   (size_t)(m0 + (tid >> 1)) * lda + k0 + 32, 0, 1);
            else
                __builtin_prefetch((const float*)Aptr +
                                   (size_t)(m0 + (tid >> 1)) * lda + k0 + 32, 0, 1);
            __builtin_prefetch(&W[(size_t)(k0 + 32 + (tid & 31)) * ldb + n0], 0, 1);
        }
        __syncthreads();

        Frag16 af[2], bf_[2];
        for (int mt = 0; mt < 2; ++mt)
            load_frag(af[mt], &lds_a[wm + mt * 16 + lane16][0], half);
        for (int nt = 0; nt < 2; ++nt)
            load_frag(bf_[nt], &lds_b[wn + nt * 16 + lane16][0], half);

        for (int mt = 0; mt < 2; ++mt)
            for (int nt = 0; nt < 2; ++nt)
                acc[mt][nt] = wmma_bf16(af[mt], bf_[nt], acc[mt][nt]);
    }

    // ---- epilogue: C layout = vgpr r -> row (r + 8*half), lane16 -> col ----
    for (int mt = 0; mt < 2; ++mt)
        for (int nt = 0; nt < 2; ++nt)
            for (int r = 0; r < 8; ++r) {
                const int m = m0 + wm + mt * 16 + r + 8 * half;
                const int n = n0 + wn + nt * 16 + lane16;
                const float v = acc[mt][nt][r] + bias[n];
                if (mode == 2) {
                    ((float*)Out)[(size_t)m * N + n] = v;
                } else {
                    const int b = m >> 11, s = m & (S_ - 1);
                    const int h = n >> 6,  d = n & (D_ - 1);
                    uint16_t* o = (uint16_t*)Out;
                    if (mode == 0)  // [B,H,S,D]
                        o[(((size_t)(b * H_ + h)) * S_ + s) * D_ + d] = f2bf(v);
                    else            // [B,H,D,S]
                        o[(((size_t)(b * H_ + h)) * D_ + d) * S_ + s] = f2bf(v);
                }
            }
}

// ---------------------------------------------------------------------------
// Flash attention: one block per (b, h, 64-query tile). 4 waves x 16 q-rows.
// K/V tiles staged via async global->LDS b128; online softmax; bf16 WMMA.
// ---------------------------------------------------------------------------
__global__ __launch_bounds__(128)
void attn_kernel(const uint32_t* __restrict__ qbuf,   // [B,H,S,D] bf16 as dwords
                 const uint32_t* __restrict__ kbuf,   // [B,H,S,D]
                 const uint32_t* __restrict__ vbuf,   // [B,H,D,S]
                 const float*    __restrict__ mask,   // [B,1,S,S]
                 uint16_t*       __restrict__ abuf) { // [B,S,E] bf16
    __shared__ uint32_t lds_k[64][36];      // 64 keys x 64 bf16 (32 dw) + pad
    __shared__ uint32_t lds_v[64][36];      // 64 d    x 64 keys
    __shared__ uint32_t lds_p[4][16][36];   // per wave: 16 rows x 64 bf16

    const int qt   = blockIdx.x & 31;
    const int bh   = blockIdx.x >> 5;
    const int b    = bh >> 4;
    const int h    = bh & 15;
    const int q0   = qt * 64;
    const int tid  = threadIdx.x;
    const int lane = tid & 31;
    const int wave = tid >> 5;
    const int half = lane >> 4;
    const int lane16 = lane & 15;
    const int qrow_base = q0 + wave * 16;

    // Q fragments resident for the whole block (A operand, K-dim = D = 64)
    Frag16 qf[2];
    {
        const uint4* q4 = (const uint4*)qbuf;
        const size_t rbase = ((size_t)bh * S_ + qrow_base + lane16) * 8;
        for (int kk = 0; kk < 2; ++kk) {
            *(uint4*)&qf[kk].u[0] = q4[rbase + kk * 4 + half];
            *(uint4*)&qf[kk].u[4] = q4[rbase + kk * 4 + 2 + half];
        }
    }

    float mrow[8], lrow[8];
    for (int r = 0; r < 8; ++r) { mrow[r] = -1e30f; lrow[r] = 0.0f; }
    v8f acc_o[4] = {};

    for (int kt = 0; kt < 32; ++kt) {
        const int k0 = kt * 64;
        __syncthreads();
        // ---- async-stage K and V tiles (512 x 16B each) ----
        for (int idx = tid; idx < 512; idx += 128) {
            const int r = idx >> 3, c4 = idx & 7;
            async_g2l_b128(lds_off_of(&lds_k[r][c4 * 4]),
                           &kbuf[((size_t)bh * S_ + k0 + r) * 32 + c4 * 4]);
            async_g2l_b128(lds_off_of(&lds_v[r][c4 * 4]),
                           &vbuf[((size_t)bh * D_ + r) * (S_ / 2) + (k0 >> 1) + c4 * 4]);
        }
        wait_async0();
        __syncthreads();

        // ---- scores S = Q K^T : 4 tiles of 16 keys, K-dim 64 (2 steps) ----
        v8f sc[4] = {};
        for (int nt = 0; nt < 4; ++nt)
            for (int kk = 0; kk < 2; ++kk) {
                Frag16 kf;
                load_frag(kf, &lds_k[nt * 16 + lane16][kk * 16], half);
                sc[nt] = wmma_bf16(qf[kk], kf, sc[nt]);
            }

        // ---- online softmax per row (rows stay inside a 16-lane half) ----
        float pv[4][8];
        for (int r = 0; r < 8; ++r) {
            const int q = qrow_base + r + 8 * half;
            const float* mk = mask + ((size_t)b * S_ + q) * S_ + k0;
            float s[4], rmax = -1e30f;
            for (int nt = 0; nt < 4; ++nt) {
                s[nt] = sc[nt][r] * 0.125f + mk[nt * 16 + lane16] * (-10000.0f);
                rmax = fmaxf(rmax, s[nt]);
            }
            for (int off = 1; off < 16; off <<= 1)
                rmax = fmaxf(rmax, __shfl_xor(rmax, off));
            const float mnew = fmaxf(mrow[r], rmax);
            const float corr = __expf(mrow[r] - mnew);
            float rsum = 0.0f;
            for (int nt = 0; nt < 4; ++nt) {
                const float p = __expf(s[nt] - mnew);
                pv[nt][r] = p;
                rsum += p;
            }
            for (int off = 1; off < 16; off <<= 1)
                rsum += __shfl_xor(rsum, off);
            lrow[r] = lrow[r] * corr + rsum;
            mrow[r] = mnew;
            for (int dt = 0; dt < 4; ++dt) acc_o[dt][r] *= corr;
        }

        // ---- re-layout P (C-format -> A-format) through LDS ----
        uint16_t* pp = (uint16_t*)&lds_p[wave][0][0];  // row stride 72 halves
        for (int r = 0; r < 8; ++r)
            for (int nt = 0; nt < 4; ++nt)
                pp[(r + 8 * half) * 72 + nt * 16 + lane16] = f2bf(pv[nt][r]);
        __syncthreads();

        // ---- O += P V : K-dim = 64 keys (2 steps), 4 tiles of 16 d-cols ----
        for (int kk = 0; kk < 2; ++kk) {
            Frag16 pf;
            load_frag(pf, &lds_p[wave][lane16][kk * 16], half);
            for (int dt = 0; dt < 4; ++dt) {
                Frag16 vf;
                load_frag(vf, &lds_v[dt * 16 + lane16][kk * 16], half);
                acc_o[dt] = wmma_bf16(pf, vf, acc_o[dt]);
            }
        }
    }

    // ---- normalize and write [B,S,E] bf16 ----
    for (int dt = 0; dt < 4; ++dt)
        for (int r = 0; r < 8; ++r) {
            const int q = qrow_base + r + 8 * half;
            const int d = dt * 16 + lane16;
            const float v = acc_o[dt][r] / lrow[r];
            abuf[((size_t)b * S_ + q) * E_ + h * D_ + d] = f2bf(v);
        }
}

// ---------------------------------------------------------------------------
extern "C" void kernel_launch(void* const* d_in, const int* in_sizes, int n_in,
                              void* d_out, int out_size, void* d_ws, size_t ws_size,
                              hipStream_t stream) {
    const float* queries = (const float*)d_in[0];
    const float* keys    = (const float*)d_in[1];
    const float* values  = (const float*)d_in[2];
    const float* mask    = (const float*)d_in[3];
    const float* Wq = (const float*)d_in[4];  const float* bq = (const float*)d_in[5];
    const float* Wk = (const float*)d_in[6];  const float* bk = (const float*)d_in[7];
    const float* Wv = (const float*)d_in[8];  const float* bv = (const float*)d_in[9];
    const float* Wo = (const float*)d_in[10]; const float* bo = (const float*)d_in[11];

    const size_t BUF = (size_t)B_ * H_ * S_ * D_ * sizeof(uint16_t);  // 8 MiB
    uint8_t* ws   = (uint8_t*)d_ws;
    void* qbuf = ws;
    void* kbuf = ws + BUF;
    void* vbuf = ws + 2 * BUF;
    void* abuf = ws + 3 * BUF;   // [B,S,E] bf16 (total 32 MiB of d_ws)

    const int M = B_ * S_, N = E_, K = E_;
    dim3 ggrid(N / 64, M / 64);  // (16, 64)

    gemm_bf16_kernel<false><<<ggrid, 128, 0, stream>>>(queries, Wq, bq, qbuf, M, N, K, E_, E_, 0);
    gemm_bf16_kernel<false><<<ggrid, 128, 0, stream>>>(keys,    Wk, bk, kbuf, M, N, K, E_, E_, 0);
    gemm_bf16_kernel<false><<<ggrid, 128, 0, stream>>>(values,  Wv, bv, vbuf, M, N, K, E_, E_, 1);

    attn_kernel<<<B_ * H_ * (S_ / 64), 128, 0, stream>>>(
        (const uint32_t*)qbuf, (const uint32_t*)kbuf, (const uint32_t*)vbuf,
        mask, (uint16_t*)abuf);

    gemm_bf16_kernel<true><<<ggrid, 128, 0, stream>>>(abuf, Wo, bo, d_out, M, N, K, E_, E_, 2);
}
